// Decoder_87746181857648
// MI455X (gfx1250) — compile-verified, matrix operand order checked by reading, not run on hardware
//
#include <hip/hip_runtime.h>
#include <hip/hip_bf16.h>

// MI455X (gfx1250) implementation.
// B=256, T=128, ENC=DEC=256, OUT=3.
//  - fold fc2*fc1 into W3[3][512] (no nonlinearity between) -> no total_hidden buffer
//  - precompute z2 = X @ attn2_w.T with WMMA, stored TRANSPOSED as z2t[B][ENC][T]
//    so the per-step score loop is coalesced across the T lanes of each wave
//  - persistent recurrent kernel: 16 blocks (16 batch rows each) x 512 threads (16 waves)
//    h,c live in LDS for all 128 steps; all dense layers via v_wmma_f32_16x16x32_f16
//  - score+softmax+context fused wave-locally (registers + shfl) -> 5 barriers/step
//  - hardware v_tanh_f32 for the 1.07G tanh ops (score loop + LSTM pointwise)
//  - weights pre-swizzled into WMMA B-fragment layout (contiguous v16h per lane)

typedef _Float16 f16;
typedef __attribute__((ext_vector_type(16))) _Float16 v16h;
typedef __attribute__((ext_vector_type(8)))  float    v8f;

#define TT   128
#define ENCD 256   // ENC == DEC == 256

__device__ __forceinline__ float fast_tanh(float x) {
#if __has_builtin(__builtin_amdgcn_tanhf)
    return __builtin_amdgcn_tanhf(x);
#elif __has_builtin(__builtin_amdgcn_tanh_f32)
    return __builtin_amdgcn_tanh_f32(x);
#else
    return tanhf(x);
#endif
}
__device__ __forceinline__ float sigm(float x) { return 0.5f + 0.5f * fast_tanh(0.5f * x); }

__device__ __forceinline__ v8f wmma16(v16h a, v16h b, v8f c) {
    return __builtin_amdgcn_wmma_f32_16x16x32_f16(false, a, false, b, (short)0, c, false, false);
}

// K placement inside an f16 A/B fragment (ISA 05_wmma: 16-bit A 16x32 layout):
// VGPR q<4: K = 2q + 8*half ; q>=4: K = 16 + 2(q-4) + 8*half ; pair (K, K+1)
__device__ __forceinline__ int kslot(int q) { return ((q & 4) << 2) | ((q & 3) << 1); }

// B fragment pre-swizzled in memory: block (ntile,ktile) = 512 halfs, lane-major.
__device__ __forceinline__ v16h loadB_frag(const f16* frags, int ntile, int ktile, int Kt, int lane) {
    return *(const v16h*)(frags + (((size_t)(ntile * Kt + ktile)) << 9) + (lane << 4));
}

// A fragment from f32 LDS (convert on load)
__device__ __forceinline__ v16h loadA_lds_f32(const float* base, int ld, int m, int kt, int hf) {
    union { v16h v; f16 e[16]; } a;
    const float* row = base + m * ld + kt * 32 + (hf << 3);
#pragma unroll
    for (int q = 0; q < 8; ++q) {
        int k = kslot(q);
        a.e[2 * q]     = (f16)row[k];
        a.e[2 * q + 1] = (f16)row[k + 1];
    }
    return a.v;
}

// A fragment from concatenated [h | c] f32 LDS (K = 512)
__device__ __forceinline__ v16h loadA_hc(const float* h, const float* c, int m, int kt, int hf) {
    union { v16h v; f16 e[16]; } a;
    int kb = kt * 32 + (hf << 3);
#pragma unroll
    for (int q = 0; q < 8; ++q) {
        int k = kb + kslot(q);
        const float* src = (k < ENCD) ? (h + m * ENCD + k) : (c + m * ENCD + (k - ENCD));
        a.e[2 * q]     = (f16)src[0];
        a.e[2 * q + 1] = (f16)src[1];
    }
    return a.v;
}

// A fragment from f16 LDS (32-bit packed-pair loads)
__device__ __forceinline__ v16h loadA_lds_f16(const f16* base, int ld, int m, int kt, int hf) {
    union { v16h v; unsigned u[8]; } a;
    const f16* row = base + m * ld + kt * 32 + (hf << 3);
#pragma unroll
    for (int q = 0; q < 8; ++q)
        a.u[q] = *(const unsigned*)(row + kslot(q));
    return a.v;
}

// A fragment from f32 global (convert on load)
__device__ __forceinline__ v16h loadA_glb_f32(const float* base, int ld, int row0, int m, int kt, int hf) {
    union { v16h v; f16 e[16]; } a;
    const float* row = base + (size_t)(row0 + m) * ld + kt * 32 + (hf << 3);
#pragma unroll
    for (int q = 0; q < 8; ++q) {
        int k = kslot(q);
        a.e[2 * q]     = (f16)row[k];
        a.e[2 * q + 1] = (f16)row[k + 1];
    }
    return a.v;
}

// ---------------- weight swizzle: W[n][k] (f32, row stride ld, col offset off) -> B fragments (f16)
__global__ void k_swz(f16* __restrict__ dst, const float* __restrict__ src,
                      int N, int K, int ld, int off) {
    int idx = blockIdx.x * 256 + threadIdx.x;
    if (idx >= N * K) return;
    int e    = idx & 15;
    int lane = (idx >> 4) & 31;
    int fb   = idx >> 9;
    int Kt    = K >> 5;
    int ktile = fb % Kt;
    int ntile = fb / Kt;
    int n  = lane & 15;
    int hf = lane >> 4;
    int q  = e >> 1;
    int k  = kslot(q) | (hf << 3) | (e & 1);
    dst[idx] = (f16)src[(size_t)(ntile * 16 + n) * ld + off + ktile * 32 + k];
}

// ---------------- combined bias / folded head
__global__ void k_prep(float* __restrict__ bias4, float* __restrict__ W3, float* __restrict__ b3,
                       const float* __restrict__ b_ih, const float* __restrict__ b_hh,
                       const float* __restrict__ fc1_w, const float* __restrict__ fc1_b,
                       const float* __restrict__ fc2_w, const float* __restrict__ fc2_b) {
    int tid = blockIdx.x * 256 + threadIdx.x;
    if (tid < 1024) {
        bias4[tid] = b_ih[tid] + b_hh[tid];
    } else if (tid < 1024 + 3 * 512) {
        int j2 = tid - 1024;
        int o = j2 / 512, jj = j2 % 512;
        float acc = 0.f;
        for (int d = 0; d < ENCD; ++d) acc += fc2_w[o * ENCD + d] * fc1_w[d * 512 + jj];
        W3[o * 512 + jj] = acc;
    } else if (tid < 1024 + 3 * 512 + 3) {
        int o = tid - (1024 + 3 * 512);
        float acc = fc2_b[o];
        for (int d = 0; d < ENCD; ++d) acc += fc2_w[o * ENCD + d] * fc1_b[d];
        b3[o] = acc;
    }
}

// ---------------- z2t[b][f][t] = (X @ attn2_w.T + attn2_b) transposed within each batch row
__global__ __launch_bounds__(256) void k_z2(float* __restrict__ z2t, const float* __restrict__ X,
                                            const f16* __restrict__ attn2_f,
                                            const float* __restrict__ attn2_b) {
    int tid  = threadIdx.x;
    int lane = tid & 31;
    int wg   = blockIdx.x * 8 + (tid >> 5);   // 8192 waves total
    int mtile = wg >> 2;                      // 0..2047 (16 rows of b*T+t each)
    int quad  = wg & 3;                       // 4 ntiles each
    int row0  = mtile * 16;
    int m  = lane & 15;
    int hf = lane >> 4;

    v8f acc[4] = {};
#pragma unroll
    for (int kt = 0; kt < 8; ++kt) {
        v16h a = loadA_glb_f32(X, ENCD, row0, m, kt, hf);
#pragma unroll
        for (int j = 0; j < 4; ++j) {
            v16h b = loadB_frag(attn2_f, quad * 4 + j, kt, 8, lane);
            acc[j] = wmma16(a, b, acc[j]);
        }
    }
    int n    = lane & 15;
    int brow = row0 >> 7;     // batch index (T=128, row0 % 16 == 0 so tile stays in one b)
    int t0   = row0 & 127;
#pragma unroll
    for (int j = 0; j < 4; ++j) {
        int col = (quad * 4 + j) * 16 + n;
        float bb = attn2_b[col];
        float* dst = z2t + ((size_t)brow * ENCD + col) * TT + t0;
#pragma unroll
        for (int v = 0; v < 8; ++v)
            dst[v + 8 * hf] = acc[j][v] + bb;
    }
}

// ---------------- persistent recurrent decoder: 16 blocks x 512 threads (16 waves)
__global__ __launch_bounds__(512) void k_decoder(
    const float* __restrict__ X,        // [B,T,ENC]
    const float* __restrict__ z2t,      // [B,ENC,T]
    const float* __restrict__ y_seq,    // [B,T,3]
    const f16*   __restrict__ attn1_f,  // Nt=16,Kt=16
    const float* __restrict__ attn1_b,
    const float* __restrict__ attn3_w,
    const float* __restrict__ attn3_b,
    const f16*   __restrict__ tilde_f,  // Nt=32,Kt=8 (context part, cols 3..258)
    const float* __restrict__ tilde_w,  // [512,259] (y part, cols 0..2)
    const float* __restrict__ tilde_b,
    const f16*   __restrict__ wih_f,    // Nt=64,Kt=16
    const f16*   __restrict__ whh_f,    // Nt=64,Kt=8
    const float* __restrict__ bias4,    // b_ih+b_hh
    const float* __restrict__ W3,       // [3,512] folded head
    const float* __restrict__ b3,       // [3]
    float* __restrict__ out)            // [B,T,3]
{
    __shared__ float s_h[16][ENCD];
    __shared__ float s_c[16][ENCD];
    __shared__ union { float z1[16][ENCD]; f16 ytil[16][512]; } s_u;
    __shared__ f16   s_ctxh[16][ENCD];

    const int tid  = threadIdx.x;
    const int lane = tid & 31;
    const int wid  = tid >> 5;           // 0..15 : wave id == owned batch row
    const int mA   = lane & 15;          // A-fragment row
    const int hf   = lane >> 4;
    const int b0   = blockIdx.x * 16;

    // zero init h, c
    for (int i = tid; i < 16 * ENCD; i += 512) {
        s_h[0][i] = 0.f;
        s_c[0][i] = 0.f;
    }
    __syncthreads();

    // ---- step-invariant per-lane constants kept in registers for all 128 steps
    const float a3b = attn3_b[0];
    const int   nlane = lane & 15;
    const float ab1 = attn1_b[wid * 16 + nlane];
    const int   dcol = wid * 16 + nlane;
    const float bI = bias4[dcol], bF = bias4[256 + dcol],
                bG = bias4[512 + dcol], bO = bias4[768 + dcol];
    float tbp[2], tw0p[2], tw1p[2], tw2p[2];
#pragma unroll
    for (int p = 0; p < 2; ++p) {
        int col = (wid * 2 + p) * 16 + nlane;
        tbp[p]  = tilde_b[col];
        tw0p[p] = tilde_w[(size_t)col * 259 + 0];
        tw1p[p] = tilde_w[(size_t)col * 259 + 1];
        tw2p[p] = tilde_w[(size_t)col * 259 + 2];
    }
    float w3r0[16], w3r1[16], w3r2[16];
#pragma unroll
    for (int j = 0; j < 16; ++j) {
        int d = lane * 16 + j;
        w3r0[j] = W3[d];
        w3r1[j] = W3[512 + d];
        w3r2[j] = W3[1024 + d];
    }
    const float b30 = b3[0], b31 = b3[1], b32 = b3[2];

    for (int t = 0; t < TT; ++t) {
        // ---- Phase A: z1 = [h|c] @ attn1_w.T + attn1_b  (wave w -> ntile w)
        {
            v8f acc = {};
#pragma unroll
            for (int kt = 0; kt < 16; ++kt) {
                v16h a = loadA_hc(&s_h[0][0], &s_c[0][0], mA, kt, hf);
                v16h b = loadB_frag(attn1_f, wid, kt, 16, lane);
                acc = wmma16(a, b, acc);
            }
#pragma unroll
            for (int v = 0; v < 8; ++v)
                s_u.z1[v + 8 * hf][wid * 16 + nlane] = acc[v] + ab1;
        }
        __syncthreads();

        // ---- Phase B (fused, wave-local; wave w owns batch row w):
        //      scores -> softmax -> context, attention weights never leave registers
        float sc0 = a3b, sc1 = a3b, sc2 = a3b, sc3 = a3b;
        {
            const float* z1r  = s_u.z1[wid];
            const float* zrow = z2t + (size_t)(b0 + wid) * ENCD * TT;
#pragma unroll 4
            for (int f = 0; f < ENCD; ++f) {
                float z1v = z1r[f];
                float w3v = attn3_w[f];
                const float* zf = zrow + f * TT;  // 128 contiguous t values
                sc0 += w3v * fast_tanh(z1v + zf[lane]);
                sc1 += w3v * fast_tanh(z1v + zf[lane + 32]);
                sc2 += w3v * fast_tanh(z1v + zf[lane + 64]);
                sc3 += w3v * fast_tanh(z1v + zf[lane + 96]);
            }
            float mx = fmaxf(fmaxf(sc0, sc1), fmaxf(sc2, sc3));
#pragma unroll
            for (int msk = 16; msk; msk >>= 1) mx = fmaxf(mx, __shfl_xor(mx, msk, 32));
            sc0 = __expf(sc0 - mx); sc1 = __expf(sc1 - mx);
            sc2 = __expf(sc2 - mx); sc3 = __expf(sc3 - mx);
            float sum = sc0 + sc1 + sc2 + sc3;
#pragma unroll
            for (int msk = 16; msk; msk >>= 1) sum += __shfl_xor(sum, msk, 32);
            float inv = 1.f / sum;
            sc0 *= inv; sc1 *= inv; sc2 *= inv; sc3 *= inv;
        }
        // ---- Phase C: context[b][e] = sum_t att[t]*X[b][t][e]  (att via lane broadcast)
        {
            int e0 = lane * 8;
            float4 alo = {0, 0, 0, 0}, ahi = {0, 0, 0, 0};
            const float* xb = X + (size_t)(b0 + wid) * TT * ENCD + e0;
            auto ctx_acc = [&](float scv, int tbase) {
#pragma unroll 8
                for (int i = 0; i < 32; ++i) {
                    float w_ = __shfl(scv, i, 32);
                    const float4* xp = (const float4*)(xb + (size_t)(tbase + i) * ENCD);
                    float4 x0 = xp[0], x1 = xp[1];
                    alo.x += w_ * x0.x; alo.y += w_ * x0.y; alo.z += w_ * x0.z; alo.w += w_ * x0.w;
                    ahi.x += w_ * x1.x; ahi.y += w_ * x1.y; ahi.z += w_ * x1.z; ahi.w += w_ * x1.w;
                }
            };
            ctx_acc(sc0, 0); ctx_acc(sc1, 32); ctx_acc(sc2, 64); ctx_acc(sc3, 96);
            s_ctxh[wid][e0 + 0] = (f16)alo.x; s_ctxh[wid][e0 + 1] = (f16)alo.y;
            s_ctxh[wid][e0 + 2] = (f16)alo.z; s_ctxh[wid][e0 + 3] = (f16)alo.w;
            s_ctxh[wid][e0 + 4] = (f16)ahi.x; s_ctxh[wid][e0 + 5] = (f16)ahi.y;
            s_ctxh[wid][e0 + 6] = (f16)ahi.z; s_ctxh[wid][e0 + 7] = (f16)ahi.w;
        }
        __syncthreads();

        // ---- Phase D: y_tilde = [y_t | ctx] @ tilde_w.T + tilde_b (wave w -> ntiles 2w,2w+1)
#pragma unroll
        for (int p = 0; p < 2; ++p) {
            int nt = wid * 2 + p;
            v8f acc = {};
#pragma unroll
            for (int kt = 0; kt < 8; ++kt) {
                v16h a = loadA_lds_f16(&s_ctxh[0][0], ENCD, mA, kt, hf);
                v16h b = loadB_frag(tilde_f, nt, kt, 8, lane);
                acc = wmma16(a, b, acc);
            }
            int col = nt * 16 + nlane;
#pragma unroll
            for (int v = 0; v < 8; ++v) {
                int mr = v + 8 * hf;
                const float* yr = y_seq + ((size_t)(b0 + mr) * TT + t) * 3;
                float val = acc[v] + tbp[p] + yr[0] * tw0p[p] + yr[1] * tw1p[p] + yr[2] * tw2p[p];
                s_u.ytil[mr][col] = (f16)val;
            }
        }
        __syncthreads();

        // ---- Phase E: gates = ytil @ w_ih.T + h @ w_hh.T   (wave w -> d-cols [16w,16w+16))
        v8f gacc[4] = {};
#pragma unroll
        for (int kt = 0; kt < 16; ++kt) {
            v16h a = loadA_lds_f16(&s_u.ytil[0][0], 512, mA, kt, hf);
#pragma unroll
            for (int g = 0; g < 4; ++g) {
                v16h b = loadB_frag(wih_f, g * 16 + wid, kt, 16, lane);
                gacc[g] = wmma16(a, b, gacc[g]);
            }
        }
#pragma unroll
        for (int kt = 0; kt < 8; ++kt) {
            v16h a = loadA_lds_f32(&s_h[0][0], ENCD, mA, kt, hf);
#pragma unroll
            for (int g = 0; g < 4; ++g) {
                v16h b = loadB_frag(whh_f, g * 16 + wid, kt, 8, lane);
                gacc[g] = wmma16(a, b, gacc[g]);
            }
        }
        __syncthreads();  // all reads of s_h done before the update writes

        // ---- Phase F: LSTM pointwise update straight from accumulators
        {
#pragma unroll
            for (int v = 0; v < 8; ++v) {
                int mr = v + 8 * hf;
                float ig = gacc[0][v] + bI;
                float fg = gacc[1][v] + bF;
                float gg = gacc[2][v] + bG;
                float og = gacc[3][v] + bO;
                float cn = sigm(fg) * s_c[mr][dcol] + sigm(ig) * fast_tanh(gg);
                float hn = sigm(og) * fast_tanh(cn);
                if (t < TT - 1) { s_c[mr][dcol] = cn; s_h[mr][dcol] = hn; }
            }
        }
        __syncthreads();

        // ---- Phase G: out[b][t][:] = W3 @ [h2 | ctx] + b3   (wave w -> row w)
        {
            float a0 = 0.f, a1 = 0.f, a2 = 0.f;
#pragma unroll
            for (int j = 0; j < 16; ++j) {
                int d = lane * 16 + j;  // 0..511
                float v_ = (d < ENCD) ? s_h[wid][d] : (float)s_ctxh[wid][d - ENCD];
                a0 += v_ * w3r0[j];
                a1 += v_ * w3r1[j];
                a2 += v_ * w3r2[j];
            }
#pragma unroll
            for (int msk = 16; msk; msk >>= 1) {
                a0 += __shfl_xor(a0, msk, 32);
                a1 += __shfl_xor(a1, msk, 32);
                a2 += __shfl_xor(a2, msk, 32);
            }
            if (lane == 0) {
                float* op = out + ((size_t)(b0 + wid) * TT + t) * 3;
                op[0] = a0 + b30;
                op[1] = a1 + b31;
                op[2] = a2 + b32;
            }
        }
    }
}

extern "C" void kernel_launch(void* const* d_in, const int* in_sizes, int n_in,
                              void* d_out, int out_size, void* d_ws, size_t ws_size,
                              hipStream_t stream) {
    const float* X        = (const float*)d_in[0];   // [256,128,256]
    const float* y_seq    = (const float*)d_in[1];   // [256,128,3]
    const float* attn1_w  = (const float*)d_in[2];   // [256,512]
    const float* attn1_b  = (const float*)d_in[3];
    const float* attn2_w  = (const float*)d_in[4];   // [256,256]
    const float* attn2_b  = (const float*)d_in[5];
    const float* attn3_w  = (const float*)d_in[6];   // [1,256]
    const float* attn3_b  = (const float*)d_in[7];
    const float* tilde_w  = (const float*)d_in[8];   // [512,259]
    const float* tilde_b  = (const float*)d_in[9];
    const float* w_ih     = (const float*)d_in[10];  // [1024,512]
    const float* w_hh     = (const float*)d_in[11];  // [1024,256]
    const float* b_ih     = (const float*)d_in[12];
    const float* b_hh     = (const float*)d_in[13];
    const float* fc1_w    = (const float*)d_in[14];  // [256,512]
    const float* fc1_b    = (const float*)d_in[15];
    const float* fc2_w    = (const float*)d_in[16];  // [3,256]
    const float* fc2_b    = (const float*)d_in[17];

    // workspace carve (all buffers fully rewritten every call -> deterministic)
    char* p = (char*)d_ws;
    auto carve = [&](size_t bytes) -> void* {
        void* r = (void*)p;
        p += (bytes + 255) & ~(size_t)255;
        return r;
    };
    float* z2t     = (float*)carve((size_t)32768 * 256 * 4);
    f16*   attn1_f = (f16*)carve((size_t)256 * 512 * 2);
    f16*   tilde_f = (f16*)carve((size_t)512 * 256 * 2);
    f16*   wih_f   = (f16*)carve((size_t)1024 * 512 * 2);
    f16*   whh_f   = (f16*)carve((size_t)1024 * 256 * 2);
    f16*   attn2_f = (f16*)carve((size_t)256 * 256 * 2);
    float* bias4   = (float*)carve(1024 * 4);
    float* W3      = (float*)carve(3 * 512 * 4);
    float* b3      = (float*)carve(3 * 4);

    // weight swizzles into WMMA B-fragment layout
    k_swz<<<(256 * 512 + 255) / 256, 256, 0, stream>>>(attn1_f, attn1_w, 256, 512, 512, 0);
    k_swz<<<(512 * 256 + 255) / 256, 256, 0, stream>>>(tilde_f, tilde_w, 512, 256, 259, 3);
    k_swz<<<(1024 * 512 + 255) / 256, 256, 0, stream>>>(wih_f, w_ih, 1024, 512, 512, 0);
    k_swz<<<(1024 * 256 + 255) / 256, 256, 0, stream>>>(whh_f, w_hh, 1024, 256, 256, 0);
    k_swz<<<(256 * 256 + 255) / 256, 256, 0, stream>>>(attn2_f, attn2_w, 256, 256, 256, 0);

    // fused biases + folded linear head
    k_prep<<<(1024 + 3 * 512 + 3 + 255) / 256, 256, 0, stream>>>(
        bias4, W3, b3, b_ih, b_hh, fc1_w, fc1_b, fc2_w, fc2_b);

    // loop-invariant z2 = attn2(X), stored [B][ENC][T]
    k_z2<<<1024, 256, 0, stream>>>(z2t, X, attn2_f, attn2_b);

    // persistent recurrent decoder
    k_decoder<<<16, 512, 0, stream>>>(X, z2t, y_seq,
                                      attn1_f, attn1_b, attn3_w, attn3_b,
                                      tilde_f, tilde_w, tilde_b,
                                      wih_f, whh_f, bias4, W3, b3,
                                      (float*)d_out);
}
